// MeanShiftStep_2370821947779
// MI455X (gfx1250) — matrix operand order repaired and checked by reading, not compiled, not run on hardware
//
#include <hip/hip_runtime.h>

typedef __attribute__((ext_vector_type(16))) _Float16 v16h;
typedef __attribute__((ext_vector_type(8)))  _Float16 v8h;
typedef __attribute__((ext_vector_type(8)))  float    v8f;

#define WAVES_PER_BLOCK 4
#define D 32

// Concatenate two 8x f16 LDS loads into one 16x f16 WMMA fragment.
static __device__ __forceinline__ v16h load_frag16(const _Float16* p0,
                                                   const _Float16* p1) {
    v8h lo = *(const v8h*)p0;
    v8h hi = *(const v8h*)p1;
    return __builtin_shufflevector(lo, hi, 0, 1, 2, 3, 4, 5, 6, 7,
                                           8, 9, 10, 11, 12, 13, 14, 15);
}

// ---------------- prologue: per-row sum of squares for points & reference ---
__global__ void msq_sumsq_kernel(const float* __restrict__ pts,
                                 const float* __restrict__ ref,
                                 float* __restrict__ ws, int N, int M) {
    int t = blockIdx.x * blockDim.x + threadIdx.x;
    const float* src; float* dst; int row;
    if (t < N)          { src = pts; dst = ws;     row = t;     }
    else if (t < N + M) { src = ref; dst = ws + N; row = t - N; }
    else return;
    const float4* p = (const float4*)(src + row * D);
    float s = 0.f;
#pragma unroll
    for (int i = 0; i < D / 4; ++i) {
        float4 v = p[i];
        s += v.x * v.x + v.y * v.y + v.z * v.z + v.w * v.w;
    }
    dst[row] = s;
}

// ---------------- main fused mean-shift step --------------------------------
__global__ void __launch_bounds__(WAVES_PER_BLOCK * 32)
msq_main_kernel(const float* __restrict__ pts, const float* __restrict__ ref,
                const float* __restrict__ x2g, const float* __restrict__ y2g,
                float* __restrict__ out, int N, int M) {
    __shared__ __align__(16) _Float16 refRow[32][32];                 // [m][d]
    __shared__ __align__(16) _Float16 refT[32][32];                   // [d][m]
    __shared__ __align__(16) _Float16 kTile[WAVES_PER_BLOCK][16][32]; // [n][m]

    const int lane = threadIdx.x & 31;
    const int wave = threadIdx.x >> 5;
    const int g    = lane >> 4;   // half-wave group 0/1
    const int c    = lane & 15;   // 0..15

    const int tile   = blockIdx.x * WAVES_PER_BLOCK + wave;
    const int n0     = tile * 16;
    const bool active = (n0 < N);

    // A-fragment of points (16x32 f16): lane row = n0+c,
    // halves h=0..7 -> d = 8g+h ; h=8..15 -> d = 16+8g+(h-8)
    v16h aP = {};
    float x2c[8];
    if (active) {
        const float* prow = pts + (size_t)(n0 + c) * D;
#pragma unroll
        for (int h = 0; h < 8; ++h) aP[h]     = (_Float16)prow[8 * g + h];
#pragma unroll
        for (int h = 0; h < 8; ++h) aP[8 + h] = (_Float16)prow[16 + 8 * g + h];
#pragma unroll
        for (int r = 0; r < 8; ++r) x2c[r] = x2g[n0 + r + 8 * g];
    } else {
#pragma unroll
        for (int r = 0; r < 8; ++r) x2c[r] = 0.f;
    }

    v8f nomA = {}; v8f nomB = {};
    float dsum[8];
#pragma unroll
    for (int r = 0; r < 8; ++r) dsum[r] = 0.f;

    // exp(s * -1/(2*8^2)) == exp2(s * C2)
    const float C2 = -1.4426950408889634f / 128.0f;

    const int srow = threadIdx.x >> 2;        // staging row 0..31
    const int sq   = (threadIdx.x & 3) * 8;   // staging d offset

    for (int m0 = 0; m0 < M; m0 += 32) {
        __syncthreads();  // previous chunk's LDS reads are done

        // ---- stage ref[m0..m0+31][0..31] -> LDS f16, both orientations ----
        {
            const float* rrow = ref + (size_t)(m0 + srow) * D + sq;
            v8h hv;
#pragma unroll
            for (int i = 0; i < 8; ++i) hv[i] = (_Float16)rrow[i];
            *(v8h*)&refRow[srow][sq] = hv;
#pragma unroll
            for (int i = 0; i < 8; ++i) refT[sq + i][srow] = hv[i];
        }
        __syncthreads();

        // ---- GEMM1: S = P x R^T (two 16x16 tiles over the 32-m chunk) ----
        // B 32x16 layout: lane col = m-local, halves h=0..7 -> d = 8g+h ; h=8..15 -> d = 16+8g+(h-8)
        v16h b1a = load_frag16(&refRow[c][8 * g],      &refRow[c][16 + 8 * g]);
        v16h b1b = load_frag16(&refRow[c + 16][8 * g], &refRow[c + 16][16 + 8 * g]);

        v8f zero = {};
        v8f sA = __builtin_amdgcn_wmma_f32_16x16x32_f16(false, aP, false, b1a,
                                                        (short)0, zero, false, false);
        v8f sB = __builtin_amdgcn_wmma_f32_16x16x32_f16(false, aP, false, b1b,
                                                        (short)0, zero, false, false);

        float y2a = y2g[m0 + c];
        float y2b = y2g[m0 + 16 + c];

        // ---- fused Gaussian kernel + denominator + K-tile store ----
#pragma unroll
        for (int r = 0; r < 8; ++r) {
            float ka = __builtin_amdgcn_exp2f((x2c[r] + y2a - 2.0f * sA[r]) * C2);
            float kb = __builtin_amdgcn_exp2f((x2c[r] + y2b - 2.0f * sB[r]) * C2);
            dsum[r] += ka + kb;
            kTile[wave][r + 8 * g][c]      = (_Float16)ka;
            kTile[wave][r + 8 * g][c + 16] = (_Float16)kb;
        }

        // kTile is wave-private; DS is in-order per wave — just drain stores
        asm volatile("s_wait_dscnt 0" ::: "memory");

        // ---- GEMM2: nom += K x Rchunk (d split 0..15 / 16..31) ----
        // A 16x32 layout: lane row = n-local = c, halves -> K = m pattern
        v16h kA = load_frag16(&kTile[wave][c][8 * g], &kTile[wave][c][16 + 8 * g]);

        // B 32x16 layout: lane col = d, halves h=0..7 -> m = 8g+h ; h=8..15 -> m = 16+8g+(h-8)
        v16h b2a = load_frag16(&refT[c][8 * g],      &refT[c][16 + 8 * g]);
        v16h b2b = load_frag16(&refT[c + 16][8 * g], &refT[c + 16][16 + 8 * g]);

        nomA = __builtin_amdgcn_wmma_f32_16x16x32_f16(false, kA, false, b2a,
                                                      (short)0, nomA, false, false);
        nomB = __builtin_amdgcn_wmma_f32_16x16x32_f16(false, kA, false, b2b,
                                                      (short)0, nomB, false, false);
    }

    // ---- denominator: reduce across the 16 lanes of each half-group ----
#pragma unroll
    for (int r = 0; r < 8; ++r) {
#pragma unroll
        for (int m = 1; m < 16; m <<= 1)
            dsum[r] += __shfl_xor(dsum[r], m, 32);
    }

    if (active) {
#pragma unroll
        for (int r = 0; r < 8; ++r) {
            float inv = 1.0f / dsum[r];
            int row = n0 + r + 8 * g;
            out[(size_t)row * D + c]      = nomA[r] * inv;
            out[(size_t)row * D + 16 + c] = nomB[r] * inv;
        }
    }
}

// ---------------- host launch ----------------------------------------------
extern "C" void kernel_launch(void* const* d_in, const int* in_sizes, int n_in,
                              void* d_out, int out_size, void* d_ws, size_t ws_size,
                              hipStream_t stream) {
    const float* pts = (const float*)d_in[0];
    const float* ref = (const float*)d_in[1];
    float* out = (float*)d_out;
    int N = in_sizes[0] / D;
    int M = in_sizes[1] / D;

    float* ws = (float*)d_ws;   // [N] x^2 then [M] y^2  (64 KB total)

    int threads = 256;
    int blocks1 = (N + M + threads - 1) / threads;
    msq_sumsq_kernel<<<blocks1, threads, 0, stream>>>(pts, ref, ws, N, M);

    int ntiles  = (N + 15) / 16;
    int blocks2 = (ntiles + WAVES_PER_BLOCK - 1) / WAVES_PER_BLOCK;
    msq_main_kernel<<<blocks2, WAVES_PER_BLOCK * 32, 0, stream>>>(
        pts, ref, ws, ws + N, out, N, M);
}
